// AttentionDecoder_39616778338570
// MI455X (gfx1250) — matrix operand order repaired and checked by reading, not compiled, not run on hardware
//
#include <hip/hip_runtime.h>
#include <hip/hip_bf16.h>
#include <cstddef>

// ---------------- constants (match reference) ----------------
#define CB   64      // batch
#define CS   32      // seq len
#define CN   196     // encoder positions
#define CE   512     // embed
#define CH   512     // hidden
#define CENC 512     // encoder feature
#define CA   512     // attention dim
#define CV   10000   // vocab

typedef __attribute__((ext_vector_type(16))) __bf16 bf16x16;
typedef __attribute__((ext_vector_type(8)))  __bf16 bf16x8;
typedef __attribute__((ext_vector_type(8)))  float  f32x8;

__device__ __forceinline__ __bf16 f2bf(float f) {
    unsigned u = __float_as_uint(f);
    unsigned r = (u + 0x7FFFu + ((u >> 16) & 1u)) >> 16;   // RNE
    unsigned short s = (unsigned short)r;
    union { unsigned short u16; __bf16 b; } cv; cv.u16 = s;
    return cv.b;
}

__device__ __forceinline__ float sigmoidf_(float x) { return 1.f / (1.f + __expf(-x)); }

// ---------------------------------------------------------------------------
// Generic bf16 WMMA GEMM:  C[M,N] = A[M,K] * W[N,K]^T + bias[N]   (fp32 out)
// One wave -> 16x64 output tile (4 accumulators, A fragment reused 4x).
// Requires: M%16==0, K%32==0, N%16==0.
// N%64 remainder handled by CLAMPING the B row pointer (loads stay in-bounds,
// extra WMMAs compute dead values) and guarding only the store -> the K loop
// is straight-line code with no EXEC manipulation (WMMA wants EXEC all-ones).
// ---------------------------------------------------------------------------
__global__ __launch_bounds__(256)
void wmma_gemm_bf16(const __bf16* __restrict__ A, const __bf16* __restrict__ W,
                    const float* __restrict__ bias, float* __restrict__ C,
                    int M, int N, int K, int ldc)
{
    int wid  = (blockIdx.x * 256 + threadIdx.x) >> 5;
    int lane = threadIdx.x & 31;
    int tilesM  = M >> 4;
    int tilesN4 = (N + 63) >> 6;
    if (wid >= tilesM * tilesN4) return;          // wave-uniform exit
    int tm = wid % tilesM;
    int tn = wid / tilesM;
    int m0 = tm << 4;
    int n0 = tn << 6;
    int grp = lane >> 4;                           // 0 or 1
    int l15 = lane & 15;

    const __bf16* arow = A + (size_t)(m0 + l15) * K;

    f32x8 acc0 = (f32x8){0.f,0.f,0.f,0.f,0.f,0.f,0.f,0.f};
    f32x8 acc1 = acc0, acc2 = acc0, acc3 = acc0;

    // clamped B row pointers (row-major W[N,K]; B fragment = W^T)
    const __bf16* wrow[4];
#pragma unroll
    for (int t = 0; t < 4; ++t) {
        int nrow = n0 + t * 16 + l15;
        if (nrow > N - 1) nrow = N - 1;            // clamp: keep loads legal, no branch in loop
        wrow[t] = W + (size_t)nrow * K + grp * 16;
    }

    for (int kk = 0; kk < K; kk += 32) {
        // A fragment (contiguous k order, matches B): lane m holds K[grp*16..+15]
        bf16x16 a = *(const bf16x16*)(arow + kk + grp * 16);

        bf16x16 b0 = *(const bf16x16*)(wrow[0] + kk);
        bf16x16 b1 = *(const bf16x16*)(wrow[1] + kk);
        bf16x16 b2 = *(const bf16x16*)(wrow[2] + kk);
        bf16x16 b3 = *(const bf16x16*)(wrow[3] + kk);

        acc0 = __builtin_amdgcn_wmma_f32_16x16x32_bf16(false, a, false, b0, (short)0, acc0, false, false);
        acc1 = __builtin_amdgcn_wmma_f32_16x16x32_bf16(false, a, false, b1, (short)0, acc1, false, false);
        acc2 = __builtin_amdgcn_wmma_f32_16x16x32_bf16(false, a, false, b2, (short)0, acc2, false, false);
        acc3 = __builtin_amdgcn_wmma_f32_16x16x32_bf16(false, a, false, b3, (short)0, acc3, false, false);
    }

    f32x8 accs[4] = {acc0, acc1, acc2, acc3};
#pragma unroll
    for (int t = 0; t < 4; ++t) {
        int nt = n0 + t * 16;
        if (nt >= N) continue;                     // uniform guard, outside hot loop
        int n = nt + l15;
        float bv = bias ? bias[n] : 0.f;
#pragma unroll
        for (int j = 0; j < 8; ++j) {
            int m = m0 + j + grp * 8;              // C/D layout: VGPR j -> row j + 8*grp
            C[(size_t)m * ldc + n] = accs[t][j] + bv;
        }
    }
}

// ---------------- prep kernels ----------------
__global__ void cvt_f32_bf16_k(const float* __restrict__ in, __bf16* __restrict__ out, int n) {
    int i = blockIdx.x * blockDim.x + threadIdx.x;
    if (i < n) out[i] = f2bf(in[i]);
}

__global__ void mean_enc_k(const float* __restrict__ enc, __bf16* __restrict__ mean_bf) {
    int i = blockIdx.x * blockDim.x + threadIdx.x;       // B*ENC
    if (i >= CB * CENC) return;
    int b = i >> 9, e = i & 511;
    const float* p = enc + (size_t)b * CN * CENC + e;
    float s = 0.f;
    for (int n = 0; n < CN; ++n) s += p[(size_t)n * CENC];
    mean_bf[i] = f2bf(s * (1.f / (float)CN));
}

__global__ void wcat_k(const float* __restrict__ W_ih, const float* __restrict__ W_hh,
                       __bf16* __restrict__ Wcat) {
    int i = blockIdx.x * blockDim.x + threadIdx.x;       // (4H)*(E+ENC+H)
    if (i >= 4 * CH * (CE + CENC + CH)) return;
    int r = i / (CE + CENC + CH), k = i % (CE + CENC + CH);
    float v = (k < CE + CENC) ? W_ih[(size_t)r * (CE + CENC) + k]
                              : W_hh[(size_t)r * CH + (k - (CE + CENC))];
    Wcat[i] = f2bf(v);
}

__global__ void biascat_k(const float* __restrict__ b_ih, const float* __restrict__ b_hh,
                          float* __restrict__ bc) {
    int i = blockIdx.x * blockDim.x + threadIdx.x;
    if (i < 4 * CH) bc[i] = b_ih[i] + b_hh[i];
}

__global__ void embed_k(const int* __restrict__ dec, const float* __restrict__ emb,
                        const float* __restrict__ pos, __bf16* __restrict__ out) {
    int i = blockIdx.x * blockDim.x + threadIdx.x;       // S*B*E, layout [(s*B+b)*E+e]
    if (i >= CS * CB * CE) return;
    int e = i & 511;
    int sb = i >> 9;
    int b = sb & 63;
    int s = sb >> 6;
    int tok = dec[b * CS + s];
    out[i] = f2bf(emb[(size_t)tok * CE + e] + pos[(size_t)s * CE + e]);
}

// ---------------- per-step kernels ----------------
// one block per batch row b; 256 threads (8 waves)
__global__ __launch_bounds__(256)
void attention_k(const float* __restrict__ u_hs,   // [B,N,A]
                 const float* __restrict__ w_ah,   // [B,A]
                 const float* __restrict__ Aa_w,   // [A]
                 const float* __restrict__ Aa_b,   // [1]
                 const float* __restrict__ enc,    // [B,N,ENC]
                 float* __restrict__ context,      // [B,ENC]
                 float* __restrict__ alpha_out,    // [B,S,N] base
                 int s)
{
    __shared__ float sc[256];
    __shared__ float red[256];
    int b = blockIdx.x;
    int tid = threadIdx.x, lane = tid & 31, wv = tid >> 5;
    const float* wah = w_ah + (size_t)b * CA;

    // scores[n] = Aa . tanh(u_hs[b,n,:] + w_ah[b,:]) + Aa_b
    for (int n = wv; n < CN; n += 8) {
        const float* u = u_hs + ((size_t)b * CN + n) * CA;
        float acc = 0.f;
        for (int k = lane; k < CA; k += 32)
            acc += Aa_w[k] * tanhf(u[k] + wah[k]);
#pragma unroll
        for (int off = 16; off; off >>= 1) acc += __shfl_xor(acc, off, 32);
        if (lane == 0) sc[n] = acc + Aa_b[0];
    }
    __syncthreads();

    // softmax over N=196
    float v = (tid < CN) ? sc[tid] : -3.4e38f;
    red[tid] = v; __syncthreads();
    for (int off = 128; off; off >>= 1) { if (tid < off) red[tid] = fmaxf(red[tid], red[tid + off]); __syncthreads(); }
    float mx = red[0]; __syncthreads();
    float ev = (tid < CN) ? __expf(v - mx) : 0.f;
    red[tid] = ev; __syncthreads();
    for (int off = 128; off; off >>= 1) { if (tid < off) red[tid] += red[tid + off]; __syncthreads(); }
    float inv = 1.f / red[0]; __syncthreads();
    if (tid < CN) {
        float a = ev * inv;
        sc[tid] = a;
        alpha_out[((size_t)b * CS + s) * CN + tid] = a;
    }
    __syncthreads();

    // context[e] = sum_n alpha[n] * enc[b,n,e]   (enc + u_hs stay L2-resident)
    for (int e = tid; e < CENC; e += 256) {
        const float* eb = enc + (size_t)b * CN * CENC + e;
        float acc = 0.f;
        for (int n = 0; n < CN; ++n) acc += sc[n] * eb[(size_t)n * CENC];
        context[(size_t)b * CENC + e] = acc;
    }
}

__global__ void assemble_xh_k(const __bf16* __restrict__ emb_s,   // [B,E] bf16 (step slice)
                              const float*  __restrict__ context, // [B,ENC]
                              const __bf16* __restrict__ h_bf,    // [B,H]
                              __bf16* __restrict__ xh)             // [B, E+ENC+H]
{
    int i = blockIdx.x * blockDim.x + threadIdx.x;
    if (i >= CB * (CE + CENC + CH)) return;
    int b = i / (CE + CENC + CH), j = i % (CE + CENC + CH);
    __bf16 v;
    if (j < CE)             v = emb_s[b * CE + j];
    else if (j < CE + CENC) v = f2bf(context[b * CENC + (j - CE)]);
    else                    v = h_bf[b * CH + (j - CE - CENC)];
    xh[i] = v;
}

__global__ void lstm_cell_k(const float* __restrict__ gates,  // [B,4H]
                            float* __restrict__ c,            // [B,H] in-place
                            __bf16* __restrict__ h_bf,        // [B,H]
                            __bf16* __restrict__ Hall,        // [B,S,H] rows b*S+s
                            int s)
{
    int i = blockIdx.x * blockDim.x + threadIdx.x;
    if (i >= CB * CH) return;
    int b = i >> 9, j = i & 511;
    const float* g = gates + (size_t)b * 4 * CH;
    float ig = g[j], fg = g[CH + j], gg = g[2 * CH + j], og = g[3 * CH + j];
    float cn = sigmoidf_(fg) * c[i] + sigmoidf_(ig) * tanhf(gg);
    float hn = sigmoidf_(og) * tanhf(cn);
    c[i] = cn;
    __bf16 hb = f2bf(hn);
    h_bf[i] = hb;
    Hall[((size_t)b * CS + s) * CH + j] = hb;
}

// ---------------------------------------------------------------------------
extern "C" void kernel_launch(void* const* d_in, const int* in_sizes, int n_in,
                              void* d_out, int out_size, void* d_ws, size_t ws_size,
                              hipStream_t stream)
{
    const float* enc   = (const float*)d_in[0];
    const int*   dec   = (const int*)  d_in[1];
    const float* emb   = (const float*)d_in[2];
    const float* pos   = (const float*)d_in[3];
    const float* Wa_w  = (const float*)d_in[4];
    const float* Wa_b  = (const float*)d_in[5];
    const float* Ua_w  = (const float*)d_in[6];
    const float* Ua_b  = (const float*)d_in[7];
    const float* Aa_w  = (const float*)d_in[8];
    const float* Aa_b  = (const float*)d_in[9];
    const float* ihw   = (const float*)d_in[10];
    const float* ihb   = (const float*)d_in[11];
    const float* icw   = (const float*)d_in[12];
    const float* icb   = (const float*)d_in[13];
    const float* W_ih  = (const float*)d_in[14];
    const float* b_ih  = (const float*)d_in[15];
    const float* W_hh  = (const float*)d_in[16];
    const float* b_hh  = (const float*)d_in[17];
    const float* fcn_w = (const float*)d_in[18];
    const float* fcn_b = (const float*)d_in[19];

    float* out_logits = (float*)d_out;                               // [B,S,V]
    float* out_alpha  = (float*)d_out + (size_t)CB * CS * CV;        // [B,S,N]

    // ---- workspace carve ----
    char* wp = (char*)d_ws;
    auto carve = [&](size_t bytes) -> void* {
        void* r = (void*)wp; wp += (bytes + 255) & ~(size_t)255; return r;
    };
    const int KX = CE + CENC + CH;                                   // 1536
    __bf16* enc_bf   = (__bf16*)carve((size_t)CB * CN * CENC * 2);
    __bf16* Ua_bf    = (__bf16*)carve((size_t)CA * CENC * 2);
    __bf16* Wa_bf    = (__bf16*)carve((size_t)CA * CH * 2);
    __bf16* ih_bf    = (__bf16*)carve((size_t)CH * CENC * 2);
    __bf16* ic_bf    = (__bf16*)carve((size_t)CH * CENC * 2);
    __bf16* fcn_bf   = (__bf16*)carve((size_t)CV * CH * 2);
    __bf16* Wcat_bf  = (__bf16*)carve((size_t)4 * CH * KX * 2);
    __bf16* embeds   = (__bf16*)carve((size_t)CS * CB * CE * 2);
    __bf16* mean_bf  = (__bf16*)carve((size_t)CB * CENC * 2);
    float*  u_hs     = (float*) carve((size_t)CB * CN * CA * 4);
    float*  h0f      = (float*) carve((size_t)CB * CH * 4);
    float*  cbuf     = (float*) carve((size_t)CB * CH * 4);
    __bf16* h_bf     = (__bf16*)carve((size_t)CB * CH * 2);
    float*  w_ah     = (float*) carve((size_t)CB * CA * 4);
    float*  context  = (float*) carve((size_t)CB * CENC * 4);
    __bf16* xh       = (__bf16*)carve((size_t)CB * KX * 2);
    float*  gates    = (float*) carve((size_t)CB * 4 * CH * 4);
    float*  biascat  = (float*) carve((size_t)4 * CH * 4);
    __bf16* Hall     = (__bf16*)carve((size_t)CB * CS * CH * 2);
    (void)ws_size; (void)n_in; (void)in_sizes; (void)out_size;

    auto cvt = [&](const float* src, __bf16* dst, size_t n) {
        cvt_f32_bf16_k<<<(unsigned)((n + 255) / 256), 256, 0, stream>>>(src, dst, (int)n);
    };
    auto gemm = [&](const __bf16* A_, const __bf16* W_, const float* bias_,
                    float* C_, int M_, int N_, int K_, int ldc_) {
        int tiles = (M_ / 16) * ((N_ + 63) / 64);
        wmma_gemm_bf16<<<(tiles + 7) / 8, 256, 0, stream>>>(A_, W_, bias_, C_, M_, N_, K_, ldc_);
    };

    // ---- prep: bf16 conversions + fused weight cat + embeddings ----
    cvt(enc,   enc_bf, (size_t)CB * CN * CENC);
    cvt(Ua_w,  Ua_bf,  (size_t)CA * CENC);
    cvt(Wa_w,  Wa_bf,  (size_t)CA * CH);
    cvt(ihw,   ih_bf,  (size_t)CH * CENC);
    cvt(icw,   ic_bf,  (size_t)CH * CENC);
    cvt(fcn_w, fcn_bf, (size_t)CV * CH);
    {
        size_t n = (size_t)4 * CH * KX;
        wcat_k<<<(unsigned)((n + 255) / 256), 256, 0, stream>>>(W_ih, W_hh, Wcat_bf);
        biascat_k<<<(4 * CH + 255) / 256, 256, 0, stream>>>(b_ih, b_hh, biascat);
        size_t m = (size_t)CS * CB * CE;
        embed_k<<<(unsigned)((m + 255) / 256), 256, 0, stream>>>(dec, emb, pos, embeds);
        mean_enc_k<<<(CB * CENC + 255) / 256, 256, 0, stream>>>(enc, mean_bf);
    }

    // h0 / c0  (64x512x512 WMMA GEMMs), h -> bf16
    gemm(mean_bf, ih_bf, ihb, h0f,  CB, CH, CENC, CH);
    gemm(mean_bf, ic_bf, icb, cbuf, CB, CH, CENC, CH);
    cvt(h0f, h_bf, (size_t)CB * CH);

    // hoisted u_hs = enc @ Ua^T + Ua_b : 12544 x 512 x 512
    gemm(enc_bf, Ua_bf, Ua_b, u_hs, CB * CN, CA, CENC, CA);

    // ---- recurrence ----
    for (int s = 0; s < CS; ++s) {
        gemm(h_bf, Wa_bf, Wa_b, w_ah, CB, CA, CH, CA);                  // w_ah = h @ Wa^T + b
        attention_k<<<CB, 256, 0, stream>>>(u_hs, w_ah, Aa_w, Aa_b, enc,
                                            context, out_alpha, s);
        {
            size_t n = (size_t)CB * KX;
            assemble_xh_k<<<(unsigned)((n + 255) / 256), 256, 0, stream>>>(
                embeds + (size_t)s * CB * CE, context, h_bf, xh);
        }
        gemm(xh, Wcat_bf, biascat, gates, CB, 4 * CH, KX, 4 * CH);      // fused ih+hh gates
        lstm_cell_k<<<(CB * CH + 255) / 256, 256, 0, stream>>>(gates, cbuf, h_bf, Hall, s);
    }

    // ---- batched vocab projection: [B*S, H] @ fcn_w^T -> [B,S,V] directly ----
    gemm(Hall, fcn_bf, fcn_b, out_logits, CB * CS, CV, CH, CV);
}
